// MambaEncoder_53584011985405
// MI455X (gfx1250) — matrix operand order repaired and checked by reading, not compiled, not run on hardware
//
#include <hip/hip_runtime.h>

// ---------------- Problem constants (from reference) ----------------
#define NLAYERS 4
#define BATCH   4
#define SEQ     2048
#define BS      (BATCH*SEQ)        // 8192 rows
#define DMODEL  768
#define DINNER  1536
#define DSTATE  128
#define NH      24
#define HD      64
#define DCONV   4
#define CONVD   1792               // DINNER + 2*DSTATE
#define DPROJ   3352               // 2*DINNER + 2*DSTATE + NH
#define NPAD    3392               // DPROJ padded up to multiple of 64

#define BK      32                 // K per WMMA step
#define LDSROW  40                 // halves per LDS B-row (32 data + 8 pad = 80B)

// ---------------- WMMA vector types (gfx1250) ----------------
typedef __bf16        v16bf __attribute__((ext_vector_type(16)));
typedef float         v8f   __attribute__((ext_vector_type(8)));
typedef unsigned int  v4u   __attribute__((ext_vector_type(4)));
typedef int           v4i   __attribute__((ext_vector_type(4)));

struct u16x16 { v4u lo, hi; };
struct F8     { float f[8]; };

#define AS1 __attribute__((address_space(1)))
#define AS3 __attribute__((address_space(3)))

#if defined(__has_builtin)
# if __has_builtin(__builtin_amdgcn_global_load_async_to_lds_b128) && \
     __has_builtin(__builtin_amdgcn_s_wait_asynccnt)
#  define USE_ASYNC_LDS 1
# endif
#endif

// fp32 -> bf16 (round to nearest even)
__device__ __forceinline__ unsigned short f2bf(float f) {
    union { float f; unsigned int u; } v; v.f = f;
    unsigned int u = v.u;
    unsigned int r = u + 0x7FFFu + ((u >> 16) & 1u);
    return (unsigned short)(r >> 16);
}

// A fragment (16x32 bf16, MxK): lane 0-15 row M=lane, K = [k..k+7] then [k+16..k+23];
// lanes 16-31 same rows, K base +8.  Caller pre-adds (lane>>4)<<3 into the pointer.
__device__ __forceinline__ v16bf ld_fragA(const unsigned short* p) {
    u16x16 t;
    t.lo = *(const v4u*)(p);
    t.hi = *(const v4u*)(p + 16);
    return __builtin_bit_cast(v16bf, t);
}
// B fragment (32x16 bf16, KxN): lane holds column n=lane&15, K = [k..k+7] then [k+16..k+23]
// (lanes 16-31: K base +8).  Source rows are contiguous
// ([N][K]-major weights, or an LDS row).  Caller pre-adds (lane>>4)<<3.
__device__ __forceinline__ v16bf ld_fragB(const unsigned short* p) {
    u16x16 t;
    t.lo = *(const v4u*)(p);
    t.hi = *(const v4u*)(p + 16);
    return __builtin_bit_cast(v16bf, t);
}

// ---------------- fp32 -> bf16 bulk convert ----------------
__global__ void cvt_bf16_kernel(const float* __restrict__ in,
                                unsigned short* __restrict__ out, int n) {
    int i = blockIdx.x * 256 + threadIdx.x;
    if (i < n) out[i] = f2bf(in[i]);
}

// ---------------- RMSNorm over DMODEL, writes bf16 or f32 ----------------
__global__ void rmsnorm_kernel(const float* __restrict__ x,
                               const float* __restrict__ w, float eps,
                               unsigned short* __restrict__ obf,
                               float* __restrict__ of32) {
    int r = blockIdx.x, tid = threadIdx.x;
    const float* xr = x + (size_t)r * DMODEL;
    float v[3]; float ss = 0.f;
#pragma unroll
    for (int j = 0; j < 3; ++j) { v[j] = xr[tid + j * 256]; ss += v[j] * v[j]; }
    __shared__ float sm[256];
    sm[tid] = ss; __syncthreads();
    for (int o = 128; o > 0; o >>= 1) { if (tid < o) sm[tid] += sm[tid + o]; __syncthreads(); }
    float rs = rsqrtf(sm[0] * (1.0f / DMODEL) + eps);
#pragma unroll
    for (int j = 0; j < 3; ++j) {
        int c = tid + j * 256;
        float val = w[c] * v[j] * rs;
        if (obf) obf[(size_t)r * DMODEL + c] = f2bf(val);
        else     of32[(size_t)r * DMODEL + c] = val;
    }
}

// ---------------- GEMM:  C[M][ldc] = A[M][K] * Bn[N][K]^T  (+ residual) ----------
// Block = 256 thr (8 waves), tile 128(M) x 64(N); each wave owns 16 rows x 64 cols.
// B tile (64 x 32 bf16) staged in LDS per k-step, double-buffered with
// GLOBAL_LOAD_ASYNC_TO_LDS_B128, shared by all 8 waves.  K is a template
// parameter so the whole pipeline (buffer parity, prefetch conditions,
// asynccnt immediates) is fully static straight-line code.
template<int K>
__global__ __launch_bounds__(256)
void gemm_bf16_wmma(const unsigned short* __restrict__ A, int lda,
                    const unsigned short* __restrict__ Bn, int ldb, int nValid,
                    float* __restrict__ C, int ldc,
                    const float* __restrict__ resid) {
    constexpr int NSTEP = K / BK;       // 24 or 48 (even)
    __shared__ __align__(16) unsigned short smB[2][64 * LDSROW];

    int tid  = threadIdx.x;
    int lane = tid & 31;
    int wv   = tid >> 5;
    int l16  = lane & 15;
    int hi   = lane >> 4;
    int mBase = blockIdx.y * 128 + wv * 16;
    int nBase = blockIdx.x * 64;

    // --- B-tile fill assignment: thread -> (row cn, 16B chunk cp) ---
    int cn = tid >> 2;                 // 0..63 : n within tile
    int cp = tid & 3;                  // 0..3  : which 16B of the 64B row
    int gn = nBase + cn; if (gn >= nValid) gn = nValid - 1;   // clamp (pad cols unread)
    const unsigned short* bsrc = Bn + (size_t)gn * ldb + cp * 8;
    unsigned short* bdst0 = &smB[0][cn * LDSROW + cp * 8];
    unsigned short* bdst1 = &smB[1][cn * LDSROW + cp * 8];

    const unsigned short* arow = A + (size_t)(mBase + l16) * lda + (hi << 3);
    int koffB = hi << 3;               // halves offset within LDS row

    v8f acc[4] = {};

#if defined(USE_ASYNC_LDS)
    // ---------- fully static async double-buffered pipeline ----------
    __builtin_amdgcn_global_load_async_to_lds_b128((AS1 v4i*)bsrc,
                                                   (AS3 v4i*)bdst0, 0, 0);
#pragma unroll
    for (int kk = 0; kk < NSTEP; ++kk) {
        const int k   = kk * BK;
        const int cur = kk & 1;
        if (kk + 1 < NSTEP) {
            unsigned short* nxt = (cur == 0) ? bdst1 : bdst0;
            __builtin_amdgcn_global_load_async_to_lds_b128(
                (AS1 v4i*)(bsrc + k + BK), (AS3 v4i*)nxt, 0, 0);
            __builtin_amdgcn_s_wait_asynccnt(1);
        } else {
            __builtin_amdgcn_s_wait_asynccnt(0);
        }
        __syncthreads();               // all waves' fill of smB[cur] landed

        v16bf fa = ld_fragA(arow + k);
#pragma unroll
        for (int nt = 0; nt < 4; ++nt) {
            v16bf fb = ld_fragB(&smB[cur][(nt * 16 + l16) * LDSROW + koffB]);
            acc[nt] = __builtin_amdgcn_wmma_f32_16x16x32_bf16(
                false, fa, false, fb, (short)0, acc[nt], false, false);
        }
        __syncthreads();               // done reading smB[cur]; free for overwrite
    }
#else
    // ---------- synchronous fallback: reg-staged single-buffer fill ----------
#pragma unroll
    for (int kk = 0; kk < NSTEP; ++kk) {
        const int k = kk * BK;
        v4u tmp = *(const v4u*)(bsrc + k);      // global -> reg (overlaps prior compute)
        __syncthreads();                        // previous reads of smB done
        *(v4u*)bdst0 = tmp;                     // reg -> LDS
        __syncthreads();                        // tile visible to all waves

        v16bf fa = ld_fragA(arow + k);
#pragma unroll
        for (int nt = 0; nt < 4; ++nt) {
            v16bf fb = ld_fragB(&smB[0][(nt * 16 + l16) * LDSROW + koffB]);
            acc[nt] = __builtin_amdgcn_wmma_f32_16x16x32_bf16(
                false, fa, false, fb, (short)0, acc[nt], false, false);
        }
    }
#endif

    // C/D layout: VGPR v -> row = v + 8*(lane>=16); col = lane&15
#pragma unroll
    for (int nt = 0; nt < 4; ++nt) {
        F8 r = __builtin_bit_cast(F8, acc[nt]);
        int col = nBase + nt * 16 + l16;
#pragma unroll
        for (int v = 0; v < 8; ++v) {
            int row = mBase + (hi << 3) + v;
            float val = r.f[v];
            if (resid) val += resid[(size_t)row * ldc + col];
            C[(size_t)row * ldc + col] = val;
        }
    }
}

// ---------------- causal depthwise conv (width 4) + bias + SiLU ----------------
__global__ void conv_silu_kernel(const float* __restrict__ zx,
                                 const float* __restrict__ cw,
                                 const float* __restrict__ cb,
                                 float* __restrict__ xbc) {
    int idx = blockIdx.x * 256 + threadIdx.x;
    if (idx >= BS * CONVD) return;
    int c  = idx % CONVD;
    int rs = idx / CONVD;
    int s  = rs % SEQ;
    int b  = rs / SEQ;
    float acc = cb[c];
#pragma unroll
    for (int k = 0; k < DCONV; ++k) {
        int sp = s - (DCONV - 1) + k;
        if (sp >= 0)
            acc += cw[c * DCONV + k] * zx[(size_t)(b * SEQ + sp) * NPAD + DINNER + c];
    }
    float sg = 1.f / (1.f + __expf(-acc));
    xbc[idx] = acc * sg;
}

// ---------------- dt = softplus(dt_raw + bias) ----------------
__global__ void dt_softplus_kernel(const float* __restrict__ zx,
                                   const float* __restrict__ dtb,
                                   float* __restrict__ dt) {
    int idx = blockIdx.x * 256 + threadIdx.x;
    if (idx >= BS * NH) return;
    int h = idx % NH, r = idx / NH;
    float v = zx[(size_t)r * NPAD + DINNER + CONVD + h] + dtb[h];
    dt[idx] = (v > 20.f) ? v : log1pf(__expf(v));
}

// ---------------- sequential SSD scan: one WG per (batch, head) ----------------
// 256 thr: thread = p*4 + ng ; holds state h[p][ng*32 .. ng*32+31] in 32 VGPRs.
__global__ __launch_bounds__(256)
void ssd_scan_kernel(const float* __restrict__ xbc, const float* __restrict__ dt,
                     const float* __restrict__ Alog, const float* __restrict__ Dsk,
                     float* __restrict__ y) {
    int bh = blockIdx.x;
    int b = bh / NH, h = bh % NH;
    int tid = threadIdx.x;
    int p = tid >> 2, ng = tid & 3, nb = ng * 32;
    float A  = -__expf(Alog[h]);
    float Dp = Dsk[h];

    __shared__ float sB[DSTATE], sC[DSTATE], red[256];
    float hst[32];
#pragma unroll
    for (int j = 0; j < 32; ++j) hst[j] = 0.f;

    for (int s = 0; s < SEQ; ++s) {
        int r = b * SEQ + s;
        const float* xrow = xbc + (size_t)r * CONVD;
        if (tid < 128) sB[tid]       = xrow[DINNER + tid];
        else           sC[tid - 128] = xrow[DINNER + DSTATE + (tid - 128)];
        float dtv = dt[r * NH + h];
        float dA  = __expf(dtv * A);
        float xv  = xrow[h * HD + p];
        __syncthreads();
        float coef = dtv * xv;
        float acc = 0.f;
#pragma unroll
        for (int j = 0; j < 32; ++j) {
            float hj = dA * hst[j] + coef * sB[nb + j];
            hst[j] = hj;
            acc += hj * sC[nb + j];
        }
        red[tid] = acc;
        __syncthreads();
        if (ng == 0) {
            float yv = red[tid] + red[tid + 1] + red[tid + 2] + red[tid + 3] + Dp * xv;
            y[(size_t)r * DINNER + h * HD + p] = yv;
        }
    }
}

// ---------------- y * silu(z) -> rmsnorm(gnorm_w, 1e-5) -> bf16 ----------------
__global__ void gate_norm_kernel(const float* __restrict__ y,
                                 const float* __restrict__ zx,
                                 const float* __restrict__ gw,
                                 unsigned short* __restrict__ obf) {
    int r = blockIdx.x, tid = threadIdx.x;
    const float* yr = y + (size_t)r * DINNER;
    const float* zr = zx + (size_t)r * NPAD;
    float v[6]; float ss = 0.f;
#pragma unroll
    for (int j = 0; j < 6; ++j) {
        int c = tid + j * 256;
        float z = zr[c];
        float sg = 1.f / (1.f + __expf(-z));
        float t = yr[c] * z * sg;
        v[j] = t; ss += t * t;
    }
    __shared__ float sm[256];
    sm[tid] = ss; __syncthreads();
    for (int o = 128; o > 0; o >>= 1) { if (tid < o) sm[tid] += sm[tid + o]; __syncthreads(); }
    float rs = rsqrtf(sm[0] * (1.0f / DINNER) + 1e-5f);
#pragma unroll
    for (int j = 0; j < 6; ++j) {
        int c = tid + j * 256;
        obf[(size_t)r * DINNER + c] = f2bf(gw[c] * v[j] * rs);
    }
}

// =============================== host side ===================================
extern "C" void kernel_launch(void* const* d_in, const int* in_sizes, int n_in,
                              void* d_out, int out_size, void* d_ws, size_t ws_size,
                              hipStream_t stream) {
    (void)in_sizes; (void)n_in; (void)out_size; (void)ws_size;
    const float* x_in  = (const float*)d_in[0];
    const float* Wi    = (const float*)d_in[1];   // [L, DPROJ, DMODEL]
    const float* cw    = (const float*)d_in[2];   // [L, CONVD, 4]
    const float* cb    = (const float*)d_in[3];   // [L, CONVD]
    const float* dtb   = (const float*)d_in[4];   // [L, NH]
    const float* Alog  = (const float*)d_in[5];   // [L, NH]
    const float* Dsk   = (const float*)d_in[6];   // [L, NH]
    const float* gw    = (const float*)d_in[7];   // [L, DINNER]
    const float* Wo    = (const float*)d_in[8];   // [L, DMODEL, DINNER]
    const float* lnw   = (const float*)d_in[9];   // [L, DMODEL]
    const float* flnw  = (const float*)d_in[10];  // [DMODEL]

    char* base = (char*)d_ws; size_t off = 0;
    auto carve = [&](size_t bytes) -> char* {
        char* p = base + off; off += (bytes + 255) & ~(size_t)255; return p;
    };
    unsigned short* wiB  = (unsigned short*)carve((size_t)NLAYERS * DPROJ * DMODEL * 2);
    unsigned short* woB  = (unsigned short*)carve((size_t)NLAYERS * DMODEL * DINNER * 2);
    float*          xb0  = (float*)carve((size_t)BS * DMODEL * 4);
    float*          xb1  = (float*)carve((size_t)BS * DMODEL * 4);
    unsigned short* hbf  = (unsigned short*)carve((size_t)BS * DMODEL * 2);
    float*          zx   = (float*)carve((size_t)BS * NPAD * 4);
    float*          xbc  = (float*)carve((size_t)BS * CONVD * 4);
    float*          dtA  = (float*)carve((size_t)BS * NH * 4);
    float*          ybuf = (float*)carve((size_t)BS * DINNER * 4);
    unsigned short* ynbf = (unsigned short*)carve((size_t)BS * DINNER * 2);

    // weights -> bf16 (deterministic, redone every call)
    {
        int n1 = NLAYERS * DPROJ * DMODEL;
        int n2 = NLAYERS * DMODEL * DINNER;
        cvt_bf16_kernel<<<(n1 + 255) / 256, 256, 0, stream>>>(Wi, wiB, n1);
        cvt_bf16_kernel<<<(n2 + 255) / 256, 256, 0, stream>>>(Wo, woB, n2);
    }

    const float* xcur = x_in;
    float* bufs[2] = { xb0, xb1 };
    for (int l = 0; l < NLAYERS; ++l) {
        // pre-norm -> bf16 activations
        rmsnorm_kernel<<<BS, 256, 0, stream>>>(xcur, lnw + (size_t)l * DMODEL, 1e-6f,
                                               hbf, nullptr);
        // in_proj: [BS,768] x [3352,768]^T -> [BS,3392(padded)]
        gemm_bf16_wmma<DMODEL><<<dim3(NPAD / 64, BS / 128), 256, 0, stream>>>(
            hbf, DMODEL, wiB + (size_t)l * DPROJ * DMODEL, DMODEL, DPROJ,
            zx, NPAD, nullptr);
        // depthwise causal conv + SiLU over xBC channels
        conv_silu_kernel<<<(BS * CONVD) / 256, 256, 0, stream>>>(
            zx, cw + (size_t)l * CONVD * DCONV, cb + (size_t)l * CONVD, xbc);
        // dt = softplus(dt + bias)
        dt_softplus_kernel<<<(BS * NH) / 256, 256, 0, stream>>>(
            zx, dtb + (size_t)l * NH, dtA);
        // sequential SSD scan, one WG per (batch, head)
        ssd_scan_kernel<<<BATCH * NH, 256, 0, stream>>>(
            xbc, dtA, Alog + (size_t)l * NH, Dsk + (size_t)l * NH, ybuf);
        // gate + group rmsnorm -> bf16
        gate_norm_kernel<<<BS, 256, 0, stream>>>(
            ybuf, zx, gw + (size_t)l * DINNER, ynbf);
        // out_proj + residual: [BS,1536] x [768,1536]^T + xcur -> xnext
        float* xnext = bufs[l & 1];
        gemm_bf16_wmma<DINNER><<<dim3(DMODEL / 64, BS / 128), 256, 0, stream>>>(
            ynbf, DINNER, woB + (size_t)l * DMODEL * DINNER, DINNER, DMODEL,
            xnext, DMODEL, xcur);
        xcur = xnext;
    }
    // final rmsnorm -> f32 output
    rmsnorm_kernel<<<BS, 256, 0, stream>>>(xcur, flnw, 1e-6f, nullptr, (float*)d_out);
}
